// nlpdo_torch_18391049962028
// MI455X (gfx1250) — matrix operand order repaired and checked by reading, not compiled, not run on hardware
//
#include <hip/hip_runtime.h>
#include <hip/hip_bf16.h>

// ---------------------------------------------------------------------------
// CDNA5 / gfx1250. wave32. All GEMMs use v_wmma_f32_16x16x32_bf16.
// Operands are pre-packed into WMMA-fragment-major layout
//   [tile][ktile][lane(32)][elem(16)]  (16 bf16 = 32B contiguous per lane)
// so every fragment is one coalesced 32-byte load per lane. GEMM k-loop is
// software-pipelined in registers (double-buffered fragments) so loads for
// step kt+1 overlap the four WMMAs of step kt.
//
// ISA fragment layouts (CDNA5 CDNA5 ISA):
//   A (16x32 bf16): m = lane&15,  K = e + 8*(lane>>4) + 8*(e>=8)
//   B (32x16 bf16): n = lane&15; this code packs A and B alike with K = e + 16*(lane>>4)
//   D (16x16 f32):  m = e + 8*(lane>>4), n = lane&15
// ---------------------------------------------------------------------------

typedef __attribute__((ext_vector_type(16))) __bf16 v16bf;
typedef __attribute__((ext_vector_type(8)))  float  v8f;

__device__ __forceinline__ unsigned short f2bf(float f) {
    unsigned int u = __float_as_uint(f);
    unsigned int r = (u + 0x7FFFu + ((u >> 16) & 1u)) >> 16;   // RNE
    return (unsigned short)r;
}

__device__ __forceinline__ v16bf load_frag(const unsigned short* p) {
    return *(const v16bf*)p;        // 32B contiguous -> 2x global_load_b128
}

// ---------------- pack A: (M,Kdim) f32 -> [Mtiles][Ktiles][32][16] bf16 ----
__global__ __launch_bounds__(256) void pack_a_kernel(const float* __restrict__ src,
                                                     unsigned short* __restrict__ dst,
                                                     int Kdim, int total) {
    int idx = blockIdx.x * 256 + threadIdx.x;
    if (idx >= total) return;
    int e    = idx & 15;
    int lane = (idx >> 4) & 31;
    int rest = idx >> 9;
    int Ktiles = Kdim >> 5;
    int kt = rest % Ktiles;
    int mt = rest / Ktiles;
    int m = mt * 16 + (lane & 15);
    int k = kt * 32 + 16 * (lane >> 4) + e;
    dst[idx] = f2bf(src[(size_t)m * Kdim + k]);
}

// ---------------- pack B: x (8,256,4096) f32 -> [b][256][8][32][16] bf16 ---
__global__ __launch_bounds__(256) void pack_b_x_kernel(const float* __restrict__ src,
                                                       unsigned short* __restrict__ dst) {
    int idx = blockIdx.x * 256 + threadIdx.x;       // total = 2^23
    int e    = idx & 15;
    int lane = (idx >> 4) & 31;
    int kt   = (idx >> 9) & 7;
    int nt   = (idx >> 12) & 255;
    int b    = idx >> 20;
    int k = kt * 32 + 16 * (lane >> 4) + e;
    int n = nt * 16 + (lane & 15);
    dst[idx] = f2bf(src[((size_t)b * 256 + k) * 4096 + n]);
}

// ---------------- packed-fragment GEMM, software-pipelined -----------------
// C[b, M, N] (f32, row-major) = A[M,K] x B[b,K,N], operands fragment-packed.
// grid: (Ntiles/4, Mtiles/8, batch); block 256 (8 waves).
// Each wave: one 16-row m-tile x four 16-col n-tiles; KT compile-time.
template <int KT>
__global__ __launch_bounds__(256) void gemm_bf16_packed_kernel(
        const unsigned short* __restrict__ Apk,   // [Mtiles][KT][32][16]
        const unsigned short* __restrict__ Bpk,   // [b][Ntiles][KT][32][16]
        float* __restrict__ C,
        int Mtiles, int Ntiles) {
    const int lane  = threadIdx.x & 31;
    const int wave  = threadIdx.x >> 5;
    const int half  = lane >> 4;
    const int lan16 = lane & 15;
    const int bb = blockIdx.z;
    const int mtile = blockIdx.y * 8 + wave;
    const int nt0 = blockIdx.x * 4;
    const int M = Mtiles * 16, N = Ntiles * 16;

    const unsigned short* Abase = Apk + ((size_t)mtile * KT) * 512 + lane * 16;
    const unsigned short* Bbase =
        Bpk + (((size_t)bb * Ntiles + nt0) * KT) * 512 + lane * 16;

    v8f acc[4];
#pragma unroll
    for (int t = 0; t < 4; ++t) acc[t] = (v8f){0.f,0.f,0.f,0.f,0.f,0.f,0.f,0.f};

    // prologue: fragments for kt = 0
    v16bf a_cur = load_frag(Abase);
    v16bf b_cur[4];
#pragma unroll
    for (int t = 0; t < 4; ++t) b_cur[t] = load_frag(Bbase + (size_t)(t * KT) * 512);

#pragma unroll
    for (int kt = 0; kt < KT; ++kt) {
        v16bf a_nxt = a_cur;
        v16bf b_nxt[4] = {b_cur[0], b_cur[1], b_cur[2], b_cur[3]};
        if (kt + 1 < KT) {            // issue next-step loads before WMMAs
            a_nxt = load_frag(Abase + (size_t)(kt + 1) * 512);
#pragma unroll
            for (int t = 0; t < 4; ++t)
                b_nxt[t] = load_frag(Bbase + (size_t)(t * KT + kt + 1) * 512);
        }
#pragma unroll
        for (int t = 0; t < 4; ++t)
            acc[t] = __builtin_amdgcn_wmma_f32_16x16x32_bf16(
                false, a_cur, false, b_cur[t], (short)0, acc[t], false, false);
        a_cur = a_nxt;
#pragma unroll
        for (int t = 0; t < 4; ++t) b_cur[t] = b_nxt[t];
    }

    const int m0 = mtile * 16;
#pragma unroll
    for (int t = 0; t < 4; ++t)
#pragma unroll
        for (int e = 0; e < 8; ++e) {
            int m = m0 + e + 8 * half;
            int n = (nt0 + t) * 16 + lan16;
            C[((size_t)bb * M + m) * N + n] = acc[t][e];
        }
}

// ---------------- GroupNorm statistics: one block per (b, group) ----------
__global__ __launch_bounds__(256) void gn_stats_kernel(const float* __restrict__ mid,
                                                       float* __restrict__ stats) {
    const int grp = blockIdx.x;            // b*16 + g
    const int b = grp >> 4, g = grp & 15;
    const float* base = mid + ((size_t)b * 128 + g * 8) * 4096;
    float s = 0.f, ss = 0.f;
    for (int i = threadIdx.x; i < 32768; i += 256) {
        float v = base[i];
        s += v; ss += v * v;
    }
    __shared__ float sh[256], sh2[256];
    sh[threadIdx.x] = s; sh2[threadIdx.x] = ss;
    __syncthreads();
    for (int o = 128; o > 0; o >>= 1) {
        if (threadIdx.x < o) {
            sh[threadIdx.x]  += sh[threadIdx.x + o];
            sh2[threadIdx.x] += sh2[threadIdx.x + o];
        }
        __syncthreads();
    }
    if (threadIdx.x == 0) {
        float mean = sh[0] * (1.f / 32768.f);
        float var  = sh2[0] * (1.f / 32768.f) - mean * mean;
        stats[2 * grp]     = mean;
        stats[2 * grp + 1] = rsqrtf(var + 1e-5f);
    }
}

// ------- GN apply + affine + relu -> xm in PACKED-B layout [b][256][4][32][16]
__global__ __launch_bounds__(256) void gn_apply_kernel(
        const float* __restrict__ mid, const float* __restrict__ stats,
        const float* __restrict__ gamma, const float* __restrict__ beta,
        const float* __restrict__ gw, const float* __restrict__ gb,
        unsigned short* __restrict__ xmp, int total) {
    int idx = blockIdx.x * 256 + threadIdx.x;
    if (idx >= total) return;
    int pix = idx & 4095;
    int c   = (idx >> 12) & 127;
    int b   = idx >> 19;
    int grp = c >> 3;
    float mean = stats[2 * (b * 16 + grp)];
    float rstd = stats[2 * (b * 16 + grp) + 1];
    float v = mid[idx];
    v = (v - mean) * rstd * gamma[grp] + beta[grp];
    v = v * gw[grp] + gb[grp];
    v = fmaxf(v, 0.f);
    // packed-B slot: Kdim=128 (Ktiles=4), N=4096 (Ntiles=256)
    int nt = pix >> 4, n = pix & 15;
    int kt = c >> 5, kr = c & 31;
    int lane = n + 16 * (kr >> 4);
    int e    = kr & 15;
    size_t slot = (((size_t)b * 256 + nt) * 4 + kt) * 512 + lane * 16 + e;
    xmp[slot] = f2bf(v);
}

// ---------------- fused conv2 + L1-normalize + dynamic 5x5 filter --------
// grid: (256 pixel tiles of 16, 8 batches); block 256 (8 waves).
// B fragments (the xm column) are loop-invariant: held in VGPRs for the
// whole block. Output written directly in packed-B layout for conv3.
__global__ __launch_bounds__(256) void dynfilter_kernel(
        const unsigned short* __restrict__ W2p,  // A-packed [100][4][32][16]
        const unsigned short* __restrict__ xmp,  // B-packed [b][256][4][32][16]
        const float* __restrict__ x,             // (8,256,64,64) f32
        unsigned short* __restrict__ outp) {     // B-packed [b][256][8][32][16]
    __shared__ float Wd[800 * 16];       // half of the dynamic filter slab
    __shared__ float scale[64];          // 4 g_local x 16 pixels

    const int tid   = threadIdx.x;
    const int lane  = tid & 31;
    const int wave  = tid >> 5;
    const int half  = lane >> 4;
    const int lan16 = lane & 15;
    const int b = blockIdx.y;
    const int nt = blockIdx.x;
    const int pix0 = nt * 16;

    // hoist the 4 B fragments (whole 128-deep xm column for these 16 pixels)
    const unsigned short* Bb = xmp + (((size_t)b * 256 + nt) * 4) * 512 + lane * 16;
    v16bf bfr[4];
#pragma unroll
    for (int kt = 0; kt < 4; ++kt) bfr[kt] = load_frag(Bb + (size_t)kt * 512);

    for (int gp = 0; gp < 2; ++gp) {
        // ---- WMMA GEMM: Wd[r][p] = (W2[gp*800 + r][:] . xm[:][p]) / 5
        for (int mt = wave; mt < 50; mt += 8) {
            const int mtile = gp * 50 + mt;            // global 16-row tile
            const unsigned short* Aa = W2p + ((size_t)mtile * 4) * 512 + lane * 16;
            if (mt + 8 < 50)
                __builtin_prefetch(Aa + (size_t)8 * 4 * 512, 0, 3);
            v8f acc = (v8f){0.f,0.f,0.f,0.f,0.f,0.f,0.f,0.f};
#pragma unroll
            for (int kt = 0; kt < 4; ++kt) {
                v16bf a = load_frag(Aa + (size_t)kt * 512);
                acc = __builtin_amdgcn_wmma_f32_16x16x32_bf16(
                    false, a, false, bfr[kt], (short)0, acc, false, false);
            }
#pragma unroll
            for (int e = 0; e < 8; ++e) {
                int mr = mt * 16 + e + 8 * half;       // local row 0..799
                Wd[mr * 16 + lan16] = acc[e] * 0.2f;
            }
        }
        __syncthreads();

        // ---- L1 scales over 200-row chunks
        if (tid < 64) {
            int gl = tid >> 4, p = tid & 15;
            float s = 0.f;
            for (int r = 0; r < 200; ++r) s += fabsf(Wd[(gl * 200 + r) * 16 + p]);
            scale[tid] = 1.f / (s + 0.01f);
        }
        __syncthreads();

        // ---- dynamic local filtering: 128 channels (this pass) x 16 pixels
        for (int t = tid; t < 2048; t += 256) {
            int p  = t & 15;
            int ci = t >> 4;                 // 0..127
            int j  = gp * 32 + (ci & 31);    // filter index 0..63
            int i4 = ci >> 5;                // 0..3
            int c  = i4 * 64 + j;            // input/output channel
            int gl = (ci & 31) >> 3;         // local group 0..3
            int d  = j & 7;
            int pix = pix0 + p;
            int y = pix >> 6, xx = pix & 63;
            const float* xc = x + ((size_t)b * 256 + c) * 4096;
            const int wbase = (gl * 200 + d * 25) * 16 + p;
            float s = 0.f;
#pragma unroll
            for (int ky = 0; ky < 5; ++ky) {
                int yy = y + ky - 2;
                if ((unsigned)yy < 64u) {
#pragma unroll
                    for (int kx = 0; kx < 5; ++kx) {
                        int xx2 = xx + kx - 2;
                        if ((unsigned)xx2 < 64u)
                            s += Wd[wbase + (ky * 5 + kx) * 16] * xc[yy * 64 + xx2];
                    }
                }
            }
            s *= scale[gl * 16 + p];
            // write in packed-B layout for conv3 (Kdim=256 -> Ktiles=8)
            int kt = c >> 5, kr = c & 31;
            int plane = p + 16 * (kr >> 4);
            int e     = kr & 15;
            size_t slot = (((size_t)b * 256 + nt) * 8 + kt) * 512 + plane * 16 + e;
            outp[slot] = f2bf(s);
        }
        __syncthreads();
    }
}

// ---------------------------------------------------------------------------
extern "C" void kernel_launch(void* const* d_in, const int* in_sizes, int n_in,
                              void* d_out, int out_size, void* d_ws, size_t ws_size,
                              hipStream_t stream) {
    const float* x     = (const float*)d_in[0];   // (8,256,64,64)
    const float* W1    = (const float*)d_in[1];   // (128,256)
    const float* gamma = (const float*)d_in[2];   // (16)
    const float* beta  = (const float*)d_in[3];   // (16)
    const float* gw    = (const float*)d_in[4];   // (16)
    const float* gb    = (const float*)d_in[5];   // (16)
    const float* W2    = (const float*)d_in[6];   // (1600,128)
    const float* W3    = (const float*)d_in[7];   // (256,256)
    float* out = (float*)d_out;                   // (8,256,64,64)

    char* ws = (char*)d_ws;
    size_t off = 0;
    auto alloc = [&](size_t bytes) {
        char* p = ws + off;
        off += (bytes + 255) & ~(size_t)255;
        return p;
    };
    unsigned short* xbp  = (unsigned short*)alloc((size_t)8 * 256 * 4096 * 2);  // B-packed x
    unsigned short* W1p  = (unsigned short*)alloc((size_t)128 * 256 * 2);       // A-packed
    unsigned short* W2p  = (unsigned short*)alloc((size_t)1600 * 128 * 2);      // A-packed
    unsigned short* W3p  = (unsigned short*)alloc((size_t)256 * 256 * 2);       // A-packed
    float*          mid  = (float*)alloc((size_t)8 * 128 * 4096 * 4);
    float*          stats= (float*)alloc((size_t)128 * 2 * 4);
    unsigned short* xmp  = (unsigned short*)alloc((size_t)8 * 128 * 4096 * 2);  // B-packed xm
    unsigned short* outp = (unsigned short*)alloc((size_t)8 * 256 * 4096 * 2);  // B-packed
    (void)ws_size; (void)n_in; (void)in_sizes; (void)out_size;

    // 1) pack operands to fragment-major bf16
    {
        int n = 8 * 256 * 4096;
        pack_b_x_kernel<<<n / 256, 256, 0, stream>>>(x, xbp);
        n = 128 * 256;
        pack_a_kernel<<<n / 256, 256, 0, stream>>>(W1, W1p, 256, n);
        n = 1600 * 128;
        pack_a_kernel<<<n / 256, 256, 0, stream>>>(W2, W2p, 128, n);
        n = 256 * 256;
        pack_a_kernel<<<n / 256, 256, 0, stream>>>(W3, W3p, 256, n);
    }

    // 2) conv1: mid(8,128,4096) = W1(128x256) x x    [Mtiles=8, KT=8, Ntiles=256]
    gemm_bf16_packed_kernel<8><<<dim3(64, 1, 8), 256, 0, stream>>>(W1p, xbp, mid, 8, 256);

    // 3) GroupNorm stats + apply (+ReLU) -> packed-B xm
    gn_stats_kernel<<<128, 256, 0, stream>>>(mid, stats);
    {
        int total = 8 * 128 * 4096;
        gn_apply_kernel<<<total / 256, 256, 0, stream>>>(
            mid, stats, gamma, beta, gw, gb, xmp, total);
    }

    // 4) fused conv2 + L1 normalize + dynamic 5x5 filtering -> packed-B outp
    dynfilter_kernel<<<dim3(256, 8), 256, 0, stream>>>(W2p, xmp, x, outp);

    // 5) conv3: out(8,256,4096) = W3(256x256) x outp [Mtiles=16, KT=8, Ntiles=256]
    gemm_bf16_packed_kernel<8><<<dim3(64, 2, 8), 256, 0, stream>>>(W3p, outp, out, 16, 256);
}
